// StatisticalFeatureExtractor_2190433321528
// MI455X (gfx1250) — matrix operand order repaired and checked by reading, not compiled, not run on hardware
//
#include <hip/hip_runtime.h>
#include <hip/hip_bf16.h>

#define BATCH 16384
#define SLEN  2048
#define NFEAT 20
#define FPAD  32
#define H1    64
#define H2    128
#define EPSF  1e-8f
#define NACC  23
#define NRANK 6

typedef __attribute__((ext_vector_type(16))) _Float16 v16h;
typedef __attribute__((ext_vector_type(8)))  float    v8f;

// ---------------- block reductions (wave32, 256 threads = 8 waves) -------------
__device__ __forceinline__ float blockSum(float v, float* red, int tid) {
  for (int o = 16; o > 0; o >>= 1) v += __shfl_xor(v, o, 32);
  if ((tid & 31) == 0) red[tid >> 5] = v;
  __syncthreads();
  if (tid < 8) {
    float t = red[tid];
    for (int o = 4; o > 0; o >>= 1) t += __shfl_xor(t, o, 8);
    if (tid == 0) red[0] = t;
  }
  __syncthreads();
  float r = red[0];
  __syncthreads();
  return r;
}

__device__ __forceinline__ float blockMin(float v, float* red, int tid) {
  for (int o = 16; o > 0; o >>= 1) v = fminf(v, __shfl_xor(v, o, 32));
  if ((tid & 31) == 0) red[tid >> 5] = v;
  __syncthreads();
  if (tid < 8) {
    float t = red[tid];
    for (int o = 4; o > 0; o >>= 1) t = fminf(t, __shfl_xor(t, o, 8));
    if (tid == 0) red[0] = t;
  }
  __syncthreads();
  float r = red[0];
  __syncthreads();
  return r;
}

__device__ __forceinline__ float blockMax(float v, float* red, int tid) {
  for (int o = 16; o > 0; o >>= 1) v = fmaxf(v, __shfl_xor(v, o, 32));
  if ((tid & 31) == 0) red[tid >> 5] = v;
  __syncthreads();
  if (tid < 8) {
    float t = red[tid];
    for (int o = 4; o > 0; o >>= 1) t = fmaxf(t, __shfl_xor(t, o, 8));
    if (tid == 0) red[0] = t;
  }
  __syncthreads();
  float r = red[0];
  __syncthreads();
  return r;
}

// ------------- monotone float <-> sortable uint ------------
__device__ __forceinline__ unsigned f2key(float f) {
  unsigned u = __float_as_uint(f);
  return (u & 0x80000000u) ? ~u : (u | 0x80000000u);
}
__device__ __forceinline__ float key2f(unsigned s) {
  unsigned u = (s & 0x80000000u) ? (s & 0x7FFFFFFFu) : ~s;
  return __uint_as_float(u);
}

// parallel inclusive scan of 256 bins by one wave (8 bins/lane + wave scan)
__device__ __forceinline__ void scan256(unsigned* hh, int ln) {
  unsigned v[8]; unsigned s = 0u;
  #pragma unroll
  for (int j = 0; j < 8; ++j) { v[j] = hh[ln * 8 + j]; s += v[j]; }
  unsigned tot = s;
  #pragma unroll
  for (int o = 1; o < 32; o <<= 1) {
    unsigned t = __shfl_up(tot, o, 32);
    if (ln >= o) tot += t;
  }
  unsigned run = tot - s;
  #pragma unroll
  for (int j = 0; j < 8; ++j) { run += v[j]; hh[ln * 8 + j] = run; }
}

// ---------------- Kernel 1: per-row statistical features ----------------------
__global__ void __launch_bounds__(256)
feat_kernel(const float* __restrict__ x, _Float16* __restrict__ feats) {
  __shared__ float    row[SLEN];
  __shared__ unsigned keys[SLEN];
  __shared__ unsigned hist6[NRANK * 256];
  __shared__ unsigned selb[NRANK], sele[NRANK];
  __shared__ float    red[8];
  __shared__ float    red2[NACC * 8];
  __shared__ float    res[NACC];

  const int tid = threadIdx.x;
  const int b   = blockIdx.x;

  // vectorized load -> LDS (+ sortable keys) with sum/min/max partials
  float psum = 0.f, pmin = 3.4e38f, pmax = -3.4e38f;
  const float4* xr = (const float4*)(x + (size_t)b * SLEN);
  #pragma unroll
  for (int j = 0; j < SLEN / (4 * 256); ++j) {
    int idx = tid + j * 256;
    float4 v = xr[idx];
    int base = 4 * idx;
    row[base + 0] = v.x; keys[base + 0] = f2key(v.x);
    row[base + 1] = v.y; keys[base + 1] = f2key(v.y);
    row[base + 2] = v.z; keys[base + 2] = f2key(v.z);
    row[base + 3] = v.w; keys[base + 3] = f2key(v.w);
    psum += v.x + v.y + v.z + v.w;
    pmin = fminf(pmin, fminf(fminf(v.x, v.y), fminf(v.z, v.w)));
    pmax = fmaxf(pmax, fmaxf(fmaxf(v.x, v.y), fmaxf(v.z, v.w)));
  }
  __syncthreads();

  const float sum   = blockSum(psum, red, tid);   // mean needed by all threads
  const float min_v = blockMin(pmin, red, tid);
  const float max_v = blockMax(pmax, red, tid);
  const float mean  = sum * (1.0f / SLEN);

  // second pass: centered moments, diff stats, autocorr sums, seasonal
  float acc[NACC];
  #pragma unroll
  for (int a = 0; a < NACC; ++a) acc[a] = 0.f;
  // acc map: 0:m2 1:m3 2:m4 3:dsum 4:dsq 5:dabs
  //          6:sa1 7:sa1q 8:sb1 9:sb1q 10:cr1
  //          11:sa7 12:sa7q 13:sb7 14:sb7q 15:cr7
  //          16:sa24 17:sa24q 18:sb24 19:sb24q 20:cr24
  //          21:ssum 22:ssq
  for (int i = tid; i < SLEN; i += 256) {
    float xi = row[i];
    float xc = xi - mean;
    float xc2 = xc * xc;
    acc[0] += xc2; acc[1] += xc2 * xc; acc[2] += xc2 * xc2;
    if (i >= 1) {
      float d = xi - row[i - 1];
      acc[3] += d; acc[4] += d * d; acc[5] += fabsf(d);
    }
    float xi2 = xi * xi;
    if (i < SLEN - 1)  { acc[6]  += xi; acc[7]  += xi2; acc[10] += xi * row[i + 1]; }
    if (i >= 1)        { acc[8]  += xi; acc[9]  += xi2; }
    if (i < SLEN - 7)  { acc[11] += xi; acc[12] += xi2; acc[15] += xi * row[i + 7]; }
    if (i >= 7)        { acc[13] += xi; acc[14] += xi2; }
    if (i < SLEN - 24) { acc[16] += xi; acc[17] += xi2; acc[20] += xi * row[i + 24]; }
    if (i >= 24)       { acc[18] += xi; acc[19] += xi2; }
    if ((i % 24) == 0) { acc[21] += xi; acc[22] += xi2; }
  }

  // batched block reduction: wave shuffles (no barrier), then one parallel
  // cross-wave pass for all 23 accumulators (2 barriers total)
  #pragma unroll
  for (int a = 0; a < NACC; ++a) {
    float v = acc[a];
    for (int o = 16; o > 0; o >>= 1) v += __shfl_xor(v, o, 32);
    if ((tid & 31) == 0) red2[a * 8 + (tid >> 5)] = v;
  }
  __syncthreads();
  if (tid < NACC * 8) {
    float t = red2[tid];
    for (int o = 4; o > 0; o >>= 1) t += __shfl_xor(t, o, 8);
    if ((tid & 7) == 0) res[tid >> 3] = t;
  }
  __syncthreads();

  // ---- fused 6-rank exact radix-256 selection --------------------------------
  unsigned prefix[NRANK];
  unsigned rloc[NRANK] = {511u, 512u, 1023u, 1024u, 1535u, 1536u};

  // pass 0 (bits 31:24): one shared histogram for all ranks
  hist6[tid] = 0u;
  __syncthreads();
  #pragma unroll
  for (int j = 0; j < SLEN / 256; ++j) {
    unsigned k = keys[tid + j * 256];
    atomicAdd(&hist6[(k >> 24) & 255u], 1u);
  }
  __syncthreads();
  if (tid < 32) scan256(hist6, tid);
  __syncthreads();
  {
    unsigned incl = hist6[tid];
    unsigned excl = (tid > 0) ? hist6[tid - 1] : 0u;
    #pragma unroll
    for (int q = 0; q < NRANK; ++q)
      if (excl <= rloc[q] && rloc[q] < incl) { selb[q] = (unsigned)tid; sele[q] = excl; }
  }
  __syncthreads();
  #pragma unroll
  for (int q = 0; q < NRANK; ++q) { prefix[q] = selb[q] << 24; rloc[q] -= sele[q]; }
  unsigned pmask = 0xFF000000u;
  __syncthreads();

  // passes 1..3: six histograms maintained simultaneously, scans on waves 0..5
  for (int shift = 16; shift >= 0; shift -= 8) {
    #pragma unroll
    for (int q = 0; q < NRANK; ++q) hist6[q * 256 + tid] = 0u;
    __syncthreads();
    #pragma unroll
    for (int j = 0; j < SLEN / 256; ++j) {
      unsigned k = keys[tid + j * 256];
      unsigned bin = (k >> shift) & 255u;
      unsigned kp = k & pmask;
      #pragma unroll
      for (int q = 0; q < NRANK; ++q)
        if (kp == prefix[q]) atomicAdd(&hist6[q * 256 + bin], 1u);
    }
    __syncthreads();
    {
      int wv = tid >> 5;
      if (wv < NRANK) scan256(&hist6[wv * 256], tid & 31);
    }
    __syncthreads();
    #pragma unroll
    for (int q = 0; q < NRANK; ++q) {
      unsigned incl = hist6[q * 256 + tid];
      unsigned excl = (tid > 0) ? hist6[q * 256 + tid - 1] : 0u;
      if (excl <= rloc[q] && rloc[q] < incl) { selb[q] = (unsigned)tid; sele[q] = excl; }
    }
    __syncthreads();
    #pragma unroll
    for (int q = 0; q < NRANK; ++q) { prefix[q] |= selb[q] << shift; rloc[q] -= sele[q]; }
    pmask |= 255u << shift;
    __syncthreads();
  }

  if (tid == 0) {
    const float s0511 = key2f(prefix[0]);
    const float s0512 = key2f(prefix[1]);
    const float s1023 = key2f(prefix[2]);
    const float s1024 = key2f(prefix[3]);
    const float s1535 = key2f(prefix[4]);
    const float s1536 = key2f(prefix[5]);

    const float M2 = res[0],  M3 = res[1],  M4 = res[2];
    const float DS = res[3],  DQ = res[4],  DA = res[5];
    const float A1 = res[6],  A1Q = res[7], B1 = res[8],  B1Q = res[9],  C1 = res[10];
    const float A7 = res[11], A7Q = res[12], B7 = res[13], B7Q = res[14], C7 = res[15];
    const float A24 = res[16], A24Q = res[17], B24 = res[18], B24Q = res[19], C24 = res[20];
    const float SS = res[21], SSQ = res[22];

    const float m2 = M2 * (1.0f / SLEN);
    const float m3 = M3 * (1.0f / SLEN);
    const float m4 = M4 * (1.0f / SLEN);
    const float stdv = sqrtf(m2);
    const float invd = 1.0f / (SLEN - 1);
    const float dmean = DS * invd;
    const float std_diff = sqrtf(fmaxf(DQ * invd - dmean * dmean, 0.f));
    const float trend = std_diff / (stdv + EPSF);

    auto acorr = [](float sa, float saq, float sb, float sbq, float cr, float n) {
      float am = sa / n, bm = sb / n;
      float cov = cr / n - am * bm;
      float da = sqrtf(fmaxf(saq / n - am * am, 0.f));
      float db = sqrtf(fmaxf(sbq / n - bm * bm, 0.f));
      return cov / (da * db);
    };
    const float ac1  = acorr(A1,  A1Q,  B1,  B1Q,  C1,  (float)(SLEN - 1));
    const float ac7  = acorr(A7,  A7Q,  B7,  B7Q,  C7,  (float)(SLEN - 7));
    const float ac24 = acorr(A24, A24Q, B24, B24Q, C24, (float)(SLEN - 24));

    const float cv = stdv / (fabsf(mean) + EPSF);
    const float p25    = 0.25f * s0511 + 0.75f * s0512;
    const float median = 0.5f * (s1023 + s1024);
    const float p75    = 0.75f * s1535 + 0.25f * s1536;
    const float iqr = p75 - p25;
    const float sm = SS * (1.0f / 86.0f);
    const float svar = SSQ * (1.0f / 86.0f) - sm * sm;
    const float seasonal = svar / (m2 + EPSF);
    const float skew = m3 / (m2 * stdv);
    const float kurt = m4 / (m2 * m2) - 3.0f;

    float f[NFEAT];
    f[0] = mean;   f[1] = stdv;  f[2] = min_v;   f[3] = max_v;  f[4] = median;
    f[5] = trend;  f[6] = ac1;   f[7] = ac7;     f[8] = ac24;   f[9] = cv;
    f[10] = iqr;   f[11] = seasonal; f[12] = skew; f[13] = kurt;
    f[14] = (float)SLEN; f[15] = DA; f[16] = DA * invd; f[17] = std_diff;
    f[18] = p25;   f[19] = p75;

    _Float16* out = feats + (size_t)b * FPAD;
    #pragma unroll
    for (int j = 0; j < NFEAT; ++j) {
      float v = isfinite(f[j]) ? f[j] : 0.0f;
      out[j] = (_Float16)v;
    }
    #pragma unroll
    for (int j = NFEAT; j < FPAD; ++j) out[j] = (_Float16)0.0f;
  }
}

// ---------------- Kernel 2: h = relu(feats @ W1^T + b1), WMMA f16 -------------
// grid: BATCH/16 blocks, 128 threads (4 waves, one 16x16 N-tile each)
__global__ void __launch_bounds__(128)
gemm1_kernel(const _Float16* __restrict__ feats, const float* __restrict__ W1,
             const float* __restrict__ b1, _Float16* __restrict__ h) {
  const int mtile = blockIdx.x;
  const int wave = threadIdx.x >> 5;   // N-tile 0..3
  const int lane = threadIdx.x & 31;
  const int lm = lane & 15;
  const bool hi = lane >= 16;

  // A fragment: 16x32 f16 (ISA 7.12.2 layout)
  v16h a;
  const _Float16* arow = feats + (size_t)(mtile * 16 + lm) * FPAD;
  #pragma unroll
  for (int v = 0; v < 8; ++v) {
    int kb = (v < 4 ? 2 * v : 16 + 2 * (v - 4)) + (hi ? 8 : 0);
    a[2 * v]     = arow[kb];
    a[2 * v + 1] = arow[kb + 1];
  }
  // B fragment: B[k][n] = W1[n][k] (K padded 20 -> 32 with zeros)
  const int n = wave * 16 + lm;
  const int klo = hi ? 8 : 0;
  v16h bm;
  #pragma unroll
  for (int t = 0; t < 16; ++t) {
    int k = klo + (t < 8 ? t : t + 8);
    bm[t] = (k < NFEAT) ? (_Float16)W1[n * NFEAT + k] : (_Float16)0.0f;
  }
  v8f c = {};
  c = __builtin_amdgcn_wmma_f32_16x16x32_f16(false, a, false, bm,
                                             (short)0, c, false, false);
  const float bias = b1[n];
  #pragma unroll
  for (int r = 0; r < 8; ++r) {
    int m = mtile * 16 + r + (hi ? 8 : 0);
    float v = c[r] + bias;
    v = v > 0.f ? v : 0.f;           // relu
    h[(size_t)m * H1 + n] = (_Float16)v;
  }
}

// ---------------- Kernel 3: out = h @ W2^T + b2, WMMA f16, K=64 ---------------
// grid: BATCH/16 blocks, 256 threads (8 waves, one 16x16 N-tile each)
__global__ void __launch_bounds__(256)
gemm2_kernel(const _Float16* __restrict__ h, const float* __restrict__ W2,
             const float* __restrict__ b2, float* __restrict__ out) {
  const int mtile = blockIdx.x;
  const int wave = threadIdx.x >> 5;   // N-tile 0..7
  const int lane = threadIdx.x & 31;
  const int lm = lane & 15;
  const bool hi = lane >= 16;
  const int n = wave * 16 + lm;

  v8f c = {};
  const _Float16* arow = h + (size_t)(mtile * 16 + lm) * H1;
  #pragma unroll
  for (int kk = 0; kk < 2; ++kk) {
    const int kbase = kk * 32;
    v16h a, bm;
    #pragma unroll
    for (int v = 0; v < 8; ++v) {
      int kb = kbase + (v < 4 ? 2 * v : 16 + 2 * (v - 4)) + (hi ? 8 : 0);
      a[2 * v]     = arow[kb];
      a[2 * v + 1] = arow[kb + 1];
    }
    const int klo = kbase + (hi ? 8 : 0);
    #pragma unroll
    for (int t = 0; t < 16; ++t) bm[t] = (_Float16)W2[n * H1 + klo + (t < 8 ? t : t + 8)];
    c = __builtin_amdgcn_wmma_f32_16x16x32_f16(false, a, false, bm,
                                               (short)0, c, false, false);
  }
  const float bias = b2[n];
  #pragma unroll
  for (int r = 0; r < 8; ++r) {
    int m = mtile * 16 + r + (hi ? 8 : 0);
    out[(size_t)m * H2 + n] = c[r] + bias;
  }
}

extern "C" void kernel_launch(void* const* d_in, const int* in_sizes, int n_in,
                              void* d_out, int out_size, void* d_ws, size_t ws_size,
                              hipStream_t stream) {
  const float* x  = (const float*)d_in[0];   // [16384, 2048]
  const float* W1 = (const float*)d_in[1];   // [64, 20]
  const float* b1 = (const float*)d_in[2];   // [64]
  const float* W2 = (const float*)d_in[3];   // [128, 64]
  const float* b2 = (const float*)d_in[4];   // [128]
  float* out = (float*)d_out;                // [16384, 128] f32

  _Float16* feats = (_Float16*)d_ws;                         // BATCH*FPAD f16 (1 MB)
  _Float16* hbuf  = feats + (size_t)BATCH * FPAD;            // BATCH*H1   f16 (2 MB)

  feat_kernel<<<BATCH, 256, 0, stream>>>(x, feats);
  gemm1_kernel<<<BATCH / 16, 128, 0, stream>>>(feats, W1, b1, hbuf);
  gemm2_kernel<<<BATCH / 16, 256, 0, stream>>>(hbuf, W2, b2, out);
}
